// Attention_70291434766394
// MI455X (gfx1250) — matrix operand order repaired and checked by reading, not compiled, not run on hardware
//
#include <hip/hip_runtime.h>
#include <hip/hip_bf16.h>

// ---------------------------------------------------------------------------
// Sizes (compile-time, match reference)
// ---------------------------------------------------------------------------
#define BATCH 2
#define SEQ   2048
#define DIM   2048
#define NH    16
#define NKV   4
#define HD    128
#define EPSI  1e-6f

typedef __bf16 bf16;
typedef __attribute__((ext_vector_type(16))) __bf16 v16bf;
typedef __attribute__((ext_vector_type(8)))  float  v8f;

union AFrag { v16bf v; unsigned u[8]; uint4 q[2]; };
union CFrag { v8f   v; float    f[8]; };

static __device__ __forceinline__ bf16 to_bf16(float f) {
  unsigned u = __float_as_uint(f);
  unsigned r = (u + 0x7FFFu + ((u >> 16) & 1u)) >> 16;
  return __builtin_bit_cast(__bf16, (unsigned short)r);
}
static __device__ __forceinline__ unsigned pack_bf16x2(float a, float b) {
  unsigned ua = (unsigned)__builtin_bit_cast(unsigned short, to_bf16(a));
  unsigned ub = (unsigned)__builtin_bit_cast(unsigned short, to_bf16(b));
  return ua | (ub << 16);
}
static __device__ __forceinline__ unsigned lds_off(const void* p) {
  // addrspace(3) flat address: low 32 bits are the wave-relative LDS offset
  return (unsigned)(unsigned long long)p;
}
static __device__ __forceinline__ void async_copy_b128(unsigned ldsaddr,
                                                       const void* gaddr) {
  asm volatile("global_load_async_to_lds_b128 %0, %1, off"
               :: "v"(ldsaddr), "v"((unsigned long long)gaddr) : "memory");
}
static __device__ __forceinline__ v8f wmma_bf16(const AFrag& a, const AFrag& b, v8f c) {
  // D = A(16x32 bf16) * B(32x16 bf16) + C(16x16 f32)
  return __builtin_amdgcn_wmma_f32_16x16x32_bf16(false, a.v, false, b.v,
                                                 (short)0, c, false, false);
}

// ---------------------------------------------------------------------------
// Generic tiled WMMA GEMM:  C[M,N] (f32) = A[M,K] * B[K,N](f32->bf16)
// Block: 256 thr = 8 waves (4 in M x 2 in N); block tile 128x128, K-step 32.
// Each wave: 32x64 -> 2x4 16x16 f32 accumulators. All staging loads are B128.
// ---------------------------------------------------------------------------
template <bool ABF16>
__global__ __launch_bounds__(256)
void gemm_wmma_kernel(const void* __restrict__ Araw, const float* __restrict__ B,
                      float* __restrict__ C, int M, int N, int K)
{
  __shared__ bf16 As[128][40];   // [m][k], k-padded (80B row stride, 16B aligned)
  __shared__ bf16 Bs[128][40];   // [n][k], transposed on stage-in

  const int tid  = threadIdx.x;
  const int wave = tid >> 5, lane = tid & 31;
  const int wm = wave & 3, wn = wave >> 2;
  const int r = lane & 15, half = lane >> 4;
  const int mb = blockIdx.y * 128, nb = blockIdx.x * 128;

  CFrag acc[2][4];
#pragma unroll
  for (int a = 0; a < 2; ++a)
#pragma unroll
    for (int bn = 0; bn < 4; ++bn)
#pragma unroll
      for (int j = 0; j < 8; ++j) acc[a][bn].f[j] = 0.0f;

  for (int kb = 0; kb < K; kb += 32) {
    __syncthreads();
    // ---- stage A tile (128 x 32) ----
    if constexpr (ABF16) {
      const bf16* A = (const bf16*)Araw;
#pragma unroll
      for (int i = 0; i < 2; ++i) {                 // 512 chunks of 8 bf16
        int c = i * 256 + tid;
        int m = c >> 2, k8 = (c & 3) * 8;
        *(uint4*)&As[m][k8] =
            *(const uint4*)&A[(size_t)(mb + m) * K + (kb + k8)];
      }
    } else {
      const float* A = (const float*)Araw;
#pragma unroll
      for (int i = 0; i < 4; ++i) {                 // 1024 chunks of 4 f32
        int c = i * 256 + tid;
        int m = c >> 3, k4 = (c & 7) * 4;
        float4 a4 = *(const float4*)&A[(size_t)(mb + m) * K + (kb + k4)];
        uint2 p; p.x = pack_bf16x2(a4.x, a4.y); p.y = pack_bf16x2(a4.z, a4.w);
        *(uint2*)&As[m][k4] = p;
      }
    }
    // ---- stage B tile (32 x 128) transposed to [n][k] ----
#pragma unroll
    for (int i = 0; i < 4; ++i) {                   // 1024 chunks of 4 f32
      int c = i * 256 + tid;
      int kk = c >> 5, n4 = (c & 31) * 4;
      const float* src = &B[(size_t)(kb + kk) * N + (nb + n4)];
      float4 b4 = *(const float4*)src;
      Bs[n4 + 0][kk] = to_bf16(b4.x);
      Bs[n4 + 1][kk] = to_bf16(b4.y);
      Bs[n4 + 2][kk] = to_bf16(b4.z);
      Bs[n4 + 3][kk] = to_bf16(b4.w);
      if (kb + 32 < K)                              // prefetch next B slab
        __builtin_prefetch(src + (size_t)32 * N, 0, 1);
    }
    __syncthreads();

    AFrag af[2], bf[4];
#pragma unroll
    for (int ms = 0; ms < 2; ++ms) {
      int arow = wm * 32 + ms * 16 + r;
#pragma unroll
      for (int i = 0; i < 4; ++i) {
        af[ms].u[i]     = *(const unsigned*)&As[arow][half * 8 + 2 * i];
        af[ms].u[4 + i] = *(const unsigned*)&As[arow][16 + half * 8 + 2 * i];
      }
    }
#pragma unroll
    for (int ns = 0; ns < 4; ++ns) {
      int brow = wn * 64 + ns * 16 + r;
#pragma unroll
      for (int i = 0; i < 4; ++i) {
        bf[ns].u[i]     = *(const unsigned*)&Bs[brow][half * 8 + 2 * i];
        bf[ns].u[4 + i] = *(const unsigned*)&Bs[brow][16 + half * 8 + 2 * i];
      }
    }
#pragma unroll
    for (int ms = 0; ms < 2; ++ms)
#pragma unroll
      for (int ns = 0; ns < 4; ++ns)
        acc[ms][ns].v = wmma_bf16(af[ms], bf[ns], acc[ms][ns].v);
  }

#pragma unroll
  for (int ms = 0; ms < 2; ++ms)
#pragma unroll
    for (int ns = 0; ns < 4; ++ns)
#pragma unroll
      for (int j = 0; j < 8; ++j) {
        int m = mb + wm * 32 + ms * 16 + j + 8 * half;
        int n = nb + wn * 64 + ns * 16 + r;
        C[(size_t)m * N + n] = acc[ms][ns].f[j];
      }
}

// ---------------------------------------------------------------------------
// RMSNorm + RoPE + repack to bf16.
// Grid = B*T rows; block = 256 (8 waves). Wave handles 3 head-tasks:
// tasks 0..15 = q heads, 16..19 = k heads, 20..23 = v heads (transpose-store).
// Q -> [B,H,T,HD] bf16 ; K -> [B,KV,T,HD] bf16 ; V -> [B,KV,HD,T] bf16.
// ---------------------------------------------------------------------------
__global__ __launch_bounds__(256)
void norm_rope_kernel(const float* __restrict__ q32, const float* __restrict__ k32,
                      const float* __restrict__ v32, const float* __restrict__ qs,
                      const float* __restrict__ ks, const float* __restrict__ cosb,
                      const float* __restrict__ sinb, bf16* __restrict__ Qb,
                      bf16* __restrict__ Kb, bf16* __restrict__ Vtb)
{
  const int row  = blockIdx.x;          // b*T + t
  const int b    = row >> 11;           // T = 2048
  const int t    = row & (SEQ - 1);
  const int wave = threadIdx.x >> 5, lane = threadIdx.x & 31;

  for (int it = 0; it < 3; ++it) {
    const int task = wave * 3 + it;
    if (task < NH + NKV) {              // q or k head: rmsnorm + rope
      const bool isq = task < NH;
      const int  h   = isq ? task : task - NH;
      const float* src = isq ? (q32 + (size_t)row * (NH * HD) + h * HD)
                             : (k32 + (size_t)row * (NKV * HD) + h * HD);
      const float* sc  = isq ? qs : ks;
      float4 v4 = *(const float4*)&src[lane * 4];
      float v[4] = {v4.x, v4.y, v4.z, v4.w};
      float ss = v[0]*v[0] + v[1]*v[1] + v[2]*v[2] + v[3]*v[3];
#pragma unroll
      for (int msk = 1; msk < 32; msk <<= 1) ss += __shfl_xor(ss, msk);
      const float rstd = rsqrtf(ss * (1.0f / (float)HD) + EPSI);
      float4 s4 = *(const float4*)&sc[lane * 4];
      float n[4] = {v[0]*rstd*s4.x, v[1]*rstd*s4.y, v[2]*rstd*s4.z, v[3]*rstd*s4.w};
      float4 c4 = *(const float4*)&cosb[t * HD + lane * 4];
      float4 w4 = *(const float4*)&sinb[t * HD + lane * 4];
      float o[4];
      o[0] = n[0] * c4.x - n[1] * w4.x;   // interleaved-pair RoPE
      o[1] = n[1] * c4.y + n[0] * w4.y;
      o[2] = n[2] * c4.z - n[3] * w4.z;
      o[3] = n[3] * c4.w + n[2] * w4.w;
      bf16* dst = isq ? (Qb + ((size_t)(b * NH + h) * SEQ + t) * HD)
                      : (Kb + ((size_t)(b * NKV + h) * SEQ + t) * HD);
      uint2 p; p.x = pack_bf16x2(o[0], o[1]); p.y = pack_bf16x2(o[2], o[3]);
      *(uint2*)&dst[lane * 4] = p;
    } else {                            // v head: convert + transpose
      const int h = task - (NH + NKV);
      const float* src = v32 + (size_t)row * (NKV * HD) + h * HD;
      float4 v4 = *(const float4*)&src[lane * 4];
      float v[4] = {v4.x, v4.y, v4.z, v4.w};
#pragma unroll
      for (int i = 0; i < 4; ++i) {
        int d = lane * 4 + i;
        Vtb[((size_t)(b * NKV + h) * HD + d) * SEQ + t] = to_bf16(v[i]);
      }
    }
  }
}

// ---------------------------------------------------------------------------
// Causal flash attention, bf16 WMMA, online softmax.
// Grid: (T/128, B*H). Block 256 = 8 waves; wave owns 16 q-rows.
// K tile [32 keys][HD] and V^T tile [HD][32 keys] staged into LDS with
// GLOBAL_LOAD_ASYNC_TO_LDS_B128 (ASYNCcnt); shared by all waves (same KV
// head). P restaged via per-wave LDS (s_wait_dscnt) to A-frag layout.
// ---------------------------------------------------------------------------
__global__ __launch_bounds__(256)
void flash_attn_kernel(const bf16* __restrict__ Q, const bf16* __restrict__ Kc,
                       const bf16* __restrict__ Vt, bf16* __restrict__ Oa)
{
  __shared__ bf16 kbuf[32][HD + 8];     // 272B row stride (16B aligned)
  __shared__ bf16 vtbuf[HD][32 + 8];    //  80B row stride (16B aligned)
  __shared__ bf16 pstage[8][16][40];

  const int tid  = threadIdx.x;
  const int wave = tid >> 5, lane = tid & 31;
  const int r = lane & 15, half = lane >> 4;
  const int qtile = blockIdx.x, bh = blockIdx.y;
  const int b = bh >> 4, h = bh & (NH - 1);
  const int kvh = h >> 2;                          // H/KV = 4 groups
  const int qbase = qtile * 128 + wave * 16;

  // --- Q fragments (16 rows x 128, 4 k-chunks of 32) via B128 global loads ---
  AFrag qf[4];
  {
    const uint4* q4 = (const uint4*)(Q + ((size_t)(b * NH + h) * SEQ + qbase) * HD
                                       + (size_t)r * HD);
#pragma unroll
    for (int kc = 0; kc < 4; ++kc) {
      qf[kc].q[0] = q4[(kc * 32 + half * 8) >> 3];
      qf[kc].q[1] = q4[(kc * 32 + 16 + half * 8) >> 3];
    }
  }

  CFrag oacc[8];
  float mrow[8], lrow[8];
#pragma unroll
  for (int j = 0; j < 8; ++j) { mrow[j] = -3.0e38f; lrow[j] = 0.0f; }
#pragma unroll
  for (int nt = 0; nt < 8; ++nt)
#pragma unroll
    for (int j = 0; j < 8; ++j) oacc[nt].f[j] = 0.0f;

  const bf16* kbp = Kc + (size_t)(b * NKV + kvh) * SEQ * HD;
  const bf16* vbp = Vt + (size_t)(b * NKV + kvh) * HD * SEQ;

  const int kend = qtile * 128 + 128;              // uniform causal bound
  for (int kb = 0; kb < kend; kb += 32) {
    __syncthreads();                               // WAR: prior readers done
    // ---- async DMA stage: K tile (32x128) and V^T tile (128x32), bf16 ----
#pragma unroll
    for (int i = 0; i < 2; ++i) {
      int c = i * 256 + tid;                       // 512 x 16B chunks each
      int kk = c >> 4, ho = (c & 15) * 8;
      async_copy_b128(lds_off(&kbuf[kk][ho]),
                      kbp + (size_t)(kb + kk) * HD + ho);
      int hd = c >> 2, ko = (c & 3) * 8;
      async_copy_b128(lds_off(&vtbuf[hd][ko]),
                      vbp + (size_t)hd * SEQ + (kb + ko));
    }
    asm volatile("s_wait_asynccnt 0" ::: "memory");
    __syncthreads();

    // S = Q * K^T  (16 x 32 scores per wave)
    CFrag s[2];
#pragma unroll
    for (int nt = 0; nt < 2; ++nt)
#pragma unroll
      for (int j = 0; j < 8; ++j) s[nt].f[j] = 0.0f;
#pragma unroll
    for (int kc = 0; kc < 4; ++kc) {
#pragma unroll
      for (int nt = 0; nt < 2; ++nt) {
        AFrag bfr;
        const int key = nt * 16 + r;
#pragma unroll
        for (int i = 0; i < 4; ++i) {
          bfr.u[i]     = *(const unsigned*)&kbuf[key][kc * 32 + half * 8 + 2 * i];
          bfr.u[4 + i] = *(const unsigned*)&kbuf[key][kc * 32 + 16 + half * 8 + 2 * i];
        }
        s[nt].v = wmma_bf16(qf[kc], bfr, s[nt].v);
      }
    }
    // scale + causal mask
#pragma unroll
    for (int nt = 0; nt < 2; ++nt)
#pragma unroll
      for (int j = 0; j < 8; ++j) {
        int kn = kb + nt * 16 + r;
        int qi = qbase + j + 8 * half;
        float sv = s[nt].f[j] * 0.08838834764831845f;   // 1/sqrt(128)
        s[nt].f[j] = (kn <= qi) ? sv : -3.0e38f;
      }
    // online softmax per q-row (row spread over 16 lanes of a half-wave)
#pragma unroll
    for (int j = 0; j < 8; ++j) {
      float rmax = fmaxf(s[0].f[j], s[1].f[j]);
#pragma unroll
      for (int msk = 1; msk < 16; msk <<= 1) rmax = fmaxf(rmax, __shfl_xor(rmax, msk));
      float mn   = fmaxf(mrow[j], rmax);
      float corr = __expf(mrow[j] - mn);
      mrow[j] = mn;
      float p0 = __expf(s[0].f[j] - mn);
      float p1 = __expf(s[1].f[j] - mn);
      float rsum = p0 + p1;
#pragma unroll
      for (int msk = 1; msk < 16; msk <<= 1) rsum += __shfl_xor(rsum, msk);
      lrow[j] = lrow[j] * corr + rsum;
#pragma unroll
      for (int nt = 0; nt < 8; ++nt) oacc[nt].f[j] *= corr;
      int m = j + 8 * half;
      pstage[wave][m][r]      = to_bf16(p0);
      pstage[wave][m][16 + r] = to_bf16(p1);
    }
    asm volatile("s_wait_dscnt 0" ::: "memory");     // wave-local LDS RAW fence

    // P (16x32) as A-fragment, O += P * V
    AFrag pf;
    {
      const unsigned* pw = (const unsigned*)&pstage[wave][r][0];
#pragma unroll
      for (int i = 0; i < 4; ++i) {
        pf.u[i]     = pw[(half * 8 + 2 * i) >> 1];
        pf.u[4 + i] = pw[(16 + half * 8 + 2 * i) >> 1];
      }
    }
#pragma unroll
    for (int nt = 0; nt < 8; ++nt) {
      AFrag bfr;
      const int hd = nt * 16 + r;
#pragma unroll
      for (int i = 0; i < 4; ++i) {
        bfr.u[i]     = *(const unsigned*)&vtbuf[hd][half * 8 + 2 * i];
        bfr.u[4 + i] = *(const unsigned*)&vtbuf[hd][16 + half * 8 + 2 * i];
      }
      oacc[nt].v = wmma_bf16(pf, bfr, oacc[nt].v);
    }
  }

  // epilogue: normalize, write bf16 attn output [B*T][H*HD]
#pragma unroll
  for (int j = 0; j < 8; ++j) {
    float inv = 1.0f / lrow[j];
    int qi = qbase + j + 8 * half;
    size_t rowoff = (size_t)(b * SEQ + qi) * (NH * HD) + h * HD;
#pragma unroll
    for (int nt = 0; nt < 8; ++nt)
      Oa[rowoff + nt * 16 + r] = to_bf16(oacc[nt].f[j] * inv);
  }
}

// ---------------------------------------------------------------------------
// Host-side orchestration
// ---------------------------------------------------------------------------
extern "C" void kernel_launch(void* const* d_in, const int* in_sizes, int n_in,
                              void* d_out, int out_size, void* d_ws, size_t ws_size,
                              hipStream_t stream)
{
  (void)in_sizes; (void)n_in; (void)out_size; (void)ws_size;
  const float* x   = (const float*)d_in[0];
  const float* Wq  = (const float*)d_in[1];
  const float* Wk  = (const float*)d_in[2];
  const float* Wv  = (const float*)d_in[3];
  const float* Wo  = (const float*)d_in[4];
  const float* qsc = (const float*)d_in[5];
  const float* ksc = (const float*)d_in[6];
  const float* cs  = (const float*)d_in[7];
  const float* sn  = (const float*)d_in[8];
  float* out = (float*)d_out;

  const int M = BATCH * SEQ;            // 4096
  char* ws = (char*)d_ws;
  size_t off = 0;
  float* q32 = (float*)(ws + off); off += (size_t)M * NH  * HD * 4;   // 33.5 MB
  float* k32 = (float*)(ws + off); off += (size_t)M * NKV * HD * 4;   //  8.4 MB
  float* v32 = (float*)(ws + off); off += (size_t)M * NKV * HD * 4;   //  8.4 MB
  bf16* Qb   = (bf16*)(ws + off);  off += (size_t)M * NH  * HD * 2;   // 16.8 MB
  bf16* Kb   = (bf16*)(ws + off);  off += (size_t)M * NKV * HD * 2;   //  4.2 MB
  bf16* Vtb  = (bf16*)(ws + off);  off += (size_t)M * NKV * HD * 2;   //  4.2 MB
  bf16* attn = (bf16*)q32;              // alias: q32 dead after norm_rope

  // 1-3: QKV projections (f32 in -> bf16 WMMA -> f32 out)
  gemm_wmma_kernel<false><<<dim3((NH * HD) / 128, M / 128), 256, 0, stream>>>(
      x, Wq, q32, M, NH * HD, DIM);
  gemm_wmma_kernel<false><<<dim3((NKV * HD) / 128, M / 128), 256, 0, stream>>>(
      x, Wk, k32, M, NKV * HD, DIM);
  gemm_wmma_kernel<false><<<dim3((NKV * HD) / 128, M / 128), 256, 0, stream>>>(
      x, Wv, v32, M, NKV * HD, DIM);

  // 4: RMSNorm + RoPE + bf16 repack (V transposed for PV WMMA)
  norm_rope_kernel<<<M, 256, 0, stream>>>(q32, k32, v32, qsc, ksc, cs, sn,
                                          Qb, Kb, Vtb);

  // 5: causal flash attention (async-LDS K/V staging)
  flash_attn_kernel<<<dim3(SEQ / 128, BATCH * NH), 256, 0, stream>>>(
      Qb, Kb, Vtb, attn);

  // 6: output projection -> f32 d_out
  gemm_wmma_kernel<true><<<dim3(DIM / 128, M / 128), 256, 0, stream>>>(
      attn, Wo, out, M, DIM, NH * HD);
}
